// CARAFE_13099650253019
// MI455X (gfx1250) — compile-verified, matrix operand order checked by reading, not run on hardware
//
#include <hip/hip_runtime.h>

#if defined(__has_builtin)
#  if __has_builtin(__builtin_amdgcn_global_load_async_to_lds_b128)
#    define HAVE_ASYNC_LDS 1
#  endif
#endif
#ifndef HAVE_ASYNC_LDS
#  define HAVE_ASYNC_LDS 0
#endif

typedef __attribute__((ext_vector_type(16))) __bf16 v16bf;
typedef __attribute__((ext_vector_type(8)))  __bf16 v8bf;
typedef __attribute__((ext_vector_type(8)))  float  v8f;

#if HAVE_ASYNC_LDS
// Pointee type per hipcc diagnostic: int __attribute__((vector_size(16)))
typedef int v4i_vs __attribute__((vector_size(4 * sizeof(int))));
typedef __attribute__((address_space(1))) v4i_vs* async_gptr;
typedef __attribute__((address_space(3))) v4i_vs* async_lptr;
#endif

// fp32 -> bf16 (round to nearest even), raw u16.
static __device__ __forceinline__ unsigned short f2bf(float f) {
    unsigned u = __float_as_uint(f);
    u = u + 0x7FFFu + ((u >> 16) & 1u);
    return (unsigned short)(u >> 16);
}

// v16bf WMMA operand: 8 halfs at p[0..7] and 8 at p[16..23] (16B aligned).
static __device__ __forceinline__ v16bf ldv16(const unsigned short* p) {
    v8bf lo = *(const v8bf*)(p);
    v8bf hi = *(const v8bf*)(p + 16);
    return __builtin_shufflevector(lo, hi, 0,1,2,3,4,5,6,7,8,9,10,11,12,13,14,15);
}

// ---------------------------------------------------------------------------
// Kernel 0a: repack weights to bf16.
//   wB [tap(9)][oc(64)][ic(256)]  from w1 (OIHW 64x256x3x3)
//   w2b[kc(112 pad)][oc(64)]      from w2 (100x64)
__global__ void k_prep(const float* __restrict__ w1, const float* __restrict__ w2,
                       unsigned short* __restrict__ wB, unsigned short* __restrict__ w2b) {
    int idx = blockIdx.x * 256 + threadIdx.x;
    if (idx < 9 * 64 * 256) {
        int p9 = idx / (64 * 256);
        int r  = idx - p9 * 64 * 256;
        int oc = r >> 8;
        int ic = r & 255;
        wB[idx] = f2bf(w1[(oc * 256 + ic) * 9 + p9]);
    }
    if (idx < 112 * 64) {
        int kc = idx >> 6, oc = idx & 63;
        float v = (kc < 100) ? w2[kc * 64 + oc] : 0.0f;
        w2b[idx] = f2bf(v);
    }
}

// Kernel 0b: x (NCHW fp32) -> xb (NCHW bf16), streaming convert.
__global__ void k_xcast(const float* __restrict__ x, unsigned short* __restrict__ xb) {
    int i = blockIdx.x * 256 + threadIdx.x;
    if (i < 4 * 256 * 64 * 64) xb[i] = f2bf(x[i]);
}

// Kernel 0c: im2col of xb into xA[pixel(16384)][tap(9)][ic(256)] bf16, so the
// conv GEMM's A rows are fully contiguous (async-to-LDS friendly). Writes are
// coalesced; reads are L2-resident (x = 16.7 MB << 192 MB L2).
__global__ void k_im2col(const unsigned short* __restrict__ xb,
                         unsigned short* __restrict__ xA) {
    int idx = blockIdx.x * 256 + threadIdx.x;   // < 16384*2304
    int p  = idx / 2304;
    int r  = idx - p * 2304;
    int t9 = r >> 8;
    int ic = r & 255;
    int nimg = p >> 12;
    int rem  = p & 4095;
    int y = rem >> 6, xx0 = rem & 63;
    int yy = y + (t9 / 3) - 1;
    int xx = xx0 + (t9 % 3) - 1;
    unsigned short v = 0;
    if (yy >= 0 && yy < 64 && xx >= 0 && xx < 64)
        v = xb[(((size_t)nimg * 256 + ic) << 12) + ((size_t)yy << 6) + xx];
    xA[idx] = v;
}

// ---------------------------------------------------------------------------
// Kernel 1: 3x3 conv 256->64 + ReLU as GEMM (M=16384 px, N=64, K=2304) via
// WMMA bf16. Block = 128 thr (4 waves) owns one 16-pixel M-tile; wave w owns
// oc-tile [16w,16w+16). A tile (16x32 bf16, 1 KB) is double-buffered in LDS
// and filled with GLOBAL_LOAD_ASYNC_TO_LDS_B128 (ASYNCcnt), overlapping the
// next chunk's fill with the current WMMA. B streamed per-lane from wB (L2).
__global__ __launch_bounds__(128) void k_conv3x3(
    const unsigned short* __restrict__ xA, const float* __restrict__ b1,
    const unsigned short* __restrict__ wB, unsigned short* __restrict__ feat) {
    __shared__ __align__(16) unsigned short At[2][16 * 32];

    const int tid  = threadIdx.x;
    const int lane = tid & 31;
    const int wv   = tid >> 5;
    const int p0   = blockIdx.x * 16;
    const int oc0  = wv * 16;

    const int mA   = lane & 15;
    const int kb   = (lane < 16) ? 0 : 8;
    const int ncol = lane & 15;

    // Fill roles: threads 0..63 each own one 16-byte span of the A tile.
    const int fm = tid >> 2, fq = tid & 3;
    const unsigned short* grow = xA + (size_t)(p0 + fm) * 2304 + fq * 8;
    unsigned short* l0 = &At[0][fm * 32 + fq * 8];
    unsigned short* l1 = &At[1][fm * 32 + fq * 8];

    // Prologue: chunk 0 into buffer 0.
    if (tid < 64) {
#if HAVE_ASYNC_LDS
        __builtin_amdgcn_global_load_async_to_lds_b128(
            (async_gptr)grow, (async_lptr)l0, 0, 0);
#else
        *(v8bf*)l0 = *(const v8bf*)grow;
#endif
    }

    v8f acc = {};
    for (int c = 0; c < 72; ++c) {            // K chunks: tap = c>>3, ic = (c&7)*32
        const int buf = c & 1;
#if HAVE_ASYNC_LDS
        asm volatile("s_wait_asynccnt 0x0" ::: "memory");
#endif
        __syncthreads();
        if (tid < 64 && c + 1 < 72) {         // prefetch next chunk into other buf
            const unsigned short* g = grow + (c + 1) * 32;
            unsigned short* l = buf ? l0 : l1;
#if HAVE_ASYNC_LDS
            __builtin_amdgcn_global_load_async_to_lds_b128(
                (async_gptr)g, (async_lptr)l, 0, 0);
#else
            *(v8bf*)l = *(const v8bf*)g;
#endif
        }
        v16bf a = ldv16(&At[buf][mA * 32 + kb]);
        const unsigned short* brow =
            wB + (size_t)(c >> 3) * 64 * 256 + (size_t)(oc0 + ncol) * 256 + (c & 7) * 32;
        __builtin_prefetch(brow + 32, 0, 0);
        v16bf b = ldv16(brow + kb);
        acc = __builtin_amdgcn_wmma_f32_16x16x32_bf16(
            false, a, false, b, (short)0, acc, false, false);
    }
    // bias + ReLU + store bf16 feat[pixel][oc]
    const int c = oc0 + ncol;
    const float bias = b1[c];
    #pragma unroll
    for (int v = 0; v < 8; ++v) {
        int m = (lane < 16) ? v : (v + 8);
        float val = acc[v] + bias;
        val = val > 0.0f ? val : 0.0f;
        feat[(size_t)(p0 + m) * 64 + c] = f2bf(val);
    }
}

// ---------------------------------------------------------------------------
// Kernel 2: 1x1 conv 64->100 (+bias) via WMMA, then per-pixel softmax(100).
__global__ __launch_bounds__(128) void k_kpred(
    const unsigned short* __restrict__ feat, const unsigned short* __restrict__ w2b,
    const float* __restrict__ b2, unsigned short* __restrict__ ksm) {
    __shared__ float sm[4][16][112];

    const int tid  = threadIdx.x;
    const int lane = tid & 31;
    const int wv   = tid >> 5;
    const int p0   = (blockIdx.x * 4 + wv) * 16;

    const int mA   = lane & 15;
    const int kb   = (lane < 16) ? 0 : 8;
    const int ncol = lane & 15;

    v8f acc[7] = {};
    #pragma unroll
    for (int ch = 0; ch < 2; ++ch) {
        v16bf a = ldv16(feat + (size_t)(p0 + mA) * 64 + ch * 32 + kb);
        #pragma unroll
        for (int nt = 0; nt < 7; ++nt) {
            v16bf b = ldv16(w2b + (size_t)(nt * 16 + ncol) * 64 + ch * 32 + kb);
            acc[nt] = __builtin_amdgcn_wmma_f32_16x16x32_bf16(
                false, a, false, b, (short)0, acc[nt], false, false);
        }
    }
    #pragma unroll
    for (int nt = 0; nt < 7; ++nt) {
        int c = nt * 16 + ncol;
        float bias = (c < 100) ? b2[c] : 0.0f;
        #pragma unroll
        for (int v = 0; v < 8; ++v) {
            int m = (lane < 16) ? v : (v + 8);
            sm[wv][m][c] = acc[nt][v] + bias;
        }
    }
    __syncthreads();
    if (lane < 16) {
        const int m = lane;
        const size_t p = (size_t)(p0 + m);
        float mx = -1e30f;
        for (int c = 0; c < 100; ++c) mx = fmaxf(mx, sm[wv][m][c]);
        float s = 0.0f;
        for (int c = 0; c < 100; ++c) s += __expf(sm[wv][m][c] - mx);
        float inv = 1.0f / s;
        for (int c = 0; c < 100; ++c)
            ksm[p * 100 + c] = f2bf(__expf(sm[wv][m][c] - mx) * inv);
    }
}

// ---------------------------------------------------------------------------
// Kernel 3: content-aware reassembly + pixel_shuffle as WMMA GEMM.
// One block per 4-pixel group. M = 16 = (4 px x 4 scale pos),
// K = 64 = (5 dy x 8 window-x, padded), N = 256 channels.
__global__ __launch_bounds__(128) void k_reassemble(
    const unsigned short* __restrict__ xb, const unsigned short* __restrict__ ksm,
    float* __restrict__ out) {
    __shared__ __align__(16) unsigned short Bt[256 * 64];   // [c][k] 32 KB
    __shared__ __align__(16) unsigned short At[16 * 64];    // [m][k]  2 KB

    const int tid  = threadIdx.x;
    const int lane = tid & 31;
    const int wv   = tid >> 5;
    const int pid0 = blockIdx.x * 4;
    const int nimg = pid0 >> 12;
    const int rem  = pid0 & 4095;
    const int y    = rem >> 6;
    const int xg0  = rem & 63;

    // A fill: banded softmax-kernel matrix, 8 elems/thread.
    #pragma unroll
    for (int j = 0; j < 8; ++j) {
        int e = tid * 8 + j;
        int m = e >> 6, k = e & 63;
        int i = m >> 2, s = m & 3;
        int dy = k >> 3, jx = k & 7;
        int dxl = jx - i;
        unsigned short v = 0;
        if (dy < 5 && dxl >= 0 && dxl < 5)
            v = ksm[(size_t)(pid0 + i) * 100 + s * 25 + dy * 5 + dxl];
        At[m * 64 + k] = v;
    }
    // B fill from pre-converted bf16 x: thread t covers channels 2t,2t+1.
    #pragma unroll
    for (int cc = 0; cc < 2; ++cc) {
        int c = tid * 2 + cc;
        for (int k = 0; k < 64; ++k) {
            int dy = k >> 3, jx = k & 7;
            int yy = y + dy - 2, xx = xg0 + jx - 2;
            unsigned short v = 0;
            if (dy < 5 && yy >= 0 && yy < 64 && xx >= 0 && xx < 64)
                v = xb[(((size_t)nimg * 256 + c) << 12) + ((size_t)yy << 6) + xx];
            Bt[c * 64 + k] = v;
        }
    }
    __syncthreads();

    const int mA   = lane & 15;
    const int kb   = (lane < 16) ? 0 : 8;
    const int ncol = lane & 15;

    for (int it = 0; it < 4; ++it) {
        const int ct = wv * 4 + it;
        v8f acc = {};
        #pragma unroll
        for (int ch = 0; ch < 2; ++ch) {
            v16bf a = ldv16(&At[mA * 64 + ch * 32 + kb]);
            v16bf b = ldv16(&Bt[(ct * 16 + ncol) * 64 + ch * 32 + kb]);
            acc = __builtin_amdgcn_wmma_f32_16x16x32_bf16(
                false, a, false, b, (short)0, acc, false, false);
        }
        // Emit directly into pixel-shuffled output (torch s-major semantics).
        const int c = ct * 16 + ncol;
        #pragma unroll
        for (int v = 0; v < 8; ++v) {
            int m = (lane < 16) ? v : (v + 8);
            int i = m >> 2, s = m & 3;
            int q   = s * 64 + (c >> 2);
            int row = 2 * y + ((c >> 1) & 1);
            int col = 2 * (xg0 + i) + (c & 1);
            out[(((size_t)nimg * 256 + q) << 14) + ((size_t)row << 7) + col] = acc[v];
        }
    }
}

// ---------------------------------------------------------------------------
extern "C" void kernel_launch(void* const* d_in, const int* in_sizes, int n_in,
                              void* d_out, int out_size, void* d_ws, size_t ws_size,
                              hipStream_t stream) {
    const float* x  = (const float*)d_in[0];   // (4,256,64,64)
    const float* w1 = (const float*)d_in[1];   // (64,256,3,3)
    const float* b1 = (const float*)d_in[2];   // (64,)
    const float* w2 = (const float*)d_in[3];   // (100,64,1,1)
    const float* b2 = (const float*)d_in[4];   // (100,)
    float* out = (float*)d_out;                // (4,256,128,128)

    char* ws = (char*)d_ws;
    unsigned short* wB   = (unsigned short*)(ws + 0);         // 9*64*256 bf16
    unsigned short* w2b  = (unsigned short*)(ws + 294912);    // 112*64 bf16
    unsigned short* feat = (unsigned short*)(ws + 309248);    // 16384*64 bf16
    unsigned short* ksm  = (unsigned short*)(ws + 2406400);   // 16384*100 bf16
    unsigned short* xb   = (unsigned short*)(ws + 5683200);   // 4.19M bf16
    unsigned short* xA   = (unsigned short*)(ws + 14071808);  // 16384*2304 bf16

    k_prep      <<<576,    256, 0, stream>>>(w1, w2, wB, w2b);
    k_xcast     <<<16384,  256, 0, stream>>>(x, xb);
    k_im2col    <<<147456, 256, 0, stream>>>(xb, xA);
    k_conv3x3   <<<1024,   128, 0, stream>>>(xA, b1, wB, feat);
    k_kpred     <<<256,    128, 0, stream>>>(feat, w2b, b2, ksm);
    k_reassemble<<<4096,   128, 0, stream>>>(xb, ksm, out);
}
